// MultiHeadAttention_77137612636791
// MI455X (gfx1250) — matrix operand order repaired and checked by reading, not compiled, not run on hardware
//
#include <hip/hip_runtime.h>
#include <hip/hip_bf16.h>

// ---------------------------------------------------------------------------
// Multi-head attention forward for gfx1250 (MI455X), wave32 + WMMA bf16.
//   B=4, S=2048, D_MODEL=512, H=8, D_K=D_V=64
// d_out = [ out: B*S*512 f32 | attn: B*H*S*S f32 ]
// ---------------------------------------------------------------------------

#define BB    4
#define SS    2048
#define DM    512
#define NH    8
#define DK    64
#define EPSLN 1e-5f

typedef __attribute__((ext_vector_type(16))) __bf16 bf16x16;
typedef __attribute__((ext_vector_type(8)))  float  f32x8;

union Bf16Frag { unsigned u[8]; bf16x16 v; };

struct RawA { float4 a0, a1, b0, b1; };

__device__ __forceinline__ unsigned short bf16one(float x) {
  __bf16 b = (__bf16)x;                       // native cvt, RNE
  return __builtin_bit_cast(unsigned short, b);
}

__device__ __forceinline__ f32x8 wmma_bf16(bf16x16 a, bf16x16 b, f32x8 c) {
  return __builtin_amdgcn_wmma_f32_16x16x32_bf16(
      /*neg_a=*/false, a, /*neg_b=*/false, b,
      /*c_mod=*/(short)0, c, /*reuse_a=*/false, /*reuse_b=*/false);
}

// A-frag (16x32 MxK, 16-bit layout): lane holds row m=lane%16; with h=lane/16
// VGPR0..3 pack K = 8h + {0..7}, VGPR4..7 pack K = 16 + 8h + {0..7}.
__device__ __forceinline__ RawA load_a_raw(const float* row, int kbase, int h) {
  const float4* g1 = reinterpret_cast<const float4*>(row + kbase + 8 * h);
  const float4* g2 = reinterpret_cast<const float4*>(row + kbase + 16 + 8 * h);
  RawA r;
  r.a0 = g1[0]; r.a1 = g1[1]; r.b0 = g2[0]; r.b1 = g2[1];
  return r;
}

__device__ __forceinline__ bf16x16 cvt_a(const RawA& r) {
  bf16x16 v;
  v[0]  = (__bf16)r.a0.x; v[1]  = (__bf16)r.a0.y; v[2]  = (__bf16)r.a0.z; v[3]  = (__bf16)r.a0.w;
  v[4]  = (__bf16)r.a1.x; v[5]  = (__bf16)r.a1.y; v[6]  = (__bf16)r.a1.z; v[7]  = (__bf16)r.a1.w;
  v[8]  = (__bf16)r.b0.x; v[9]  = (__bf16)r.b0.y; v[10] = (__bf16)r.b0.z; v[11] = (__bf16)r.b0.w;
  v[12] = (__bf16)r.b1.x; v[13] = (__bf16)r.b1.y; v[14] = (__bf16)r.b1.z; v[15] = (__bf16)r.b1.w;
  return v;
}

__device__ __forceinline__ bf16x16 load_a_bf16(const unsigned short* row, int kbase, int h) {
  Bf16Frag f;
  const uint4* g1 = reinterpret_cast<const uint4*>(row + kbase + 8 * h);
  const uint4* g2 = reinterpret_cast<const uint4*>(row + kbase + 16 + 8 * h);
  uint4 x = *g1, y = *g2;
  f.u[0] = x.x; f.u[1] = x.y; f.u[2] = x.z; f.u[3] = x.w;
  f.u[4] = y.x; f.u[5] = y.y; f.u[6] = y.z; f.u[7] = y.w;
  return f.v;
}

// B-frag (32x16 KxN): lane holds column n=lane%16; with h=lane/16 the lane's
// 16 bf16 are K = 8h + {0..7}, then K = 16 + 8h + {0..7} (same as A-frag).
__device__ __forceinline__ bf16x16 load_b_bf16(const unsigned short* row, int kbase, int h) {
  Bf16Frag f;
  const uint4* g1 = reinterpret_cast<const uint4*>(row + kbase + 8 * h);
  const uint4* g2 = reinterpret_cast<const uint4*>(row + kbase + 16 + 8 * h);
  uint4 x = *g1, y = *g2;
  f.u[0] = x.x; f.u[1] = x.y; f.u[2] = x.z; f.u[3] = x.w;
  f.u[4] = y.x; f.u[5] = y.y; f.u[6] = y.z; f.u[7] = y.w;
  return f.v;
}

// ---------------------------------------------------------------------------
// K0: transpose 512x512 f32 weights -> bf16 [N][K] (one-time per call)
// ---------------------------------------------------------------------------
__global__ void k_wtrans(const float* wq, const float* wk, const float* wv, const float* wo,
                         unsigned short* wqT, unsigned short* wkT,
                         unsigned short* wvT, unsigned short* woT) {
  int mat = blockIdx.y;
  const float* src = (mat == 0) ? wq : (mat == 1) ? wk : (mat == 2) ? wv : wo;
  unsigned short* dst = (mat == 0) ? wqT : (mat == 1) ? wkT : (mat == 2) ? wvT : woT;
  int idx = blockIdx.x * 256 + threadIdx.x;      // 0 .. 512*512-1
  int k = idx >> 9, n = idx & 511;
  dst[(size_t)n * DM + k] = bf16one(src[(size_t)k * DM + n]);
}

// ---------------------------------------------------------------------------
// K1: fused QKV projection.  Y = X[8192x512] @ W[512x512] -> bf16, stored
// head-major: qh/kh [B,H,S,64], v stored transposed vhT [B,H,64,S].
// Block tile 128x128, 8 waves, wave tile 64x32 (4x2 WMMA accumulators).
// ---------------------------------------------------------------------------
__global__ void __launch_bounds__(256, 1)
k_proj(const float* q, const float* k, const float* v,
       const unsigned short* wqT, const unsigned short* wkT,
       const unsigned short* wvT,
       unsigned short* qh, unsigned short* kh, unsigned short* vhT) {
  int mat = blockIdx.z;  // uniform
  const float* X = (mat == 0) ? q : (mat == 1) ? k : v;
  const unsigned short* WT = (mat == 0) ? wqT : (mat == 1) ? wkT : wvT;

  int tid = threadIdx.x, w = tid >> 5, lane = tid & 31;
  int h = lane >> 4, lm = lane & 15;
  int m0 = blockIdx.x * 128 + (w & 1) * 64;
  int n0 = blockIdx.y * 128 + (w >> 1) * 32;

  const float* xrow[4];
  const unsigned short* wrow[2];
#pragma unroll
  for (int mt = 0; mt < 4; ++mt) xrow[mt] = X + (size_t)(m0 + 16 * mt + lm) * DM;
#pragma unroll
  for (int nt = 0; nt < 2; ++nt) wrow[nt] = WT + (size_t)(n0 + 16 * nt + lm) * DM;

  f32x8 zero = {0.f, 0.f, 0.f, 0.f, 0.f, 0.f, 0.f, 0.f};
  f32x8 acc[4][2];
#pragma unroll
  for (int i = 0; i < 4; ++i)
#pragma unroll
    for (int j = 0; j < 2; ++j) acc[i][j] = zero;

  for (int kk = 0; kk < DM; kk += 32) {
    RawA ra[4];
#pragma unroll
    for (int mt = 0; mt < 4; ++mt) ra[mt] = load_a_raw(xrow[mt], kk, h);
    bf16x16 bfr[2];
#pragma unroll
    for (int nt = 0; nt < 2; ++nt) bfr[nt] = load_b_bf16(wrow[nt], kk, h);
#pragma unroll
    for (int mt = 0; mt < 4; ++mt) {
      bf16x16 afr = cvt_a(ra[mt]);
#pragma unroll
      for (int nt = 0; nt < 2; ++nt) acc[mt][nt] = wmma_bf16(afr, bfr[nt], acc[mt][nt]);
    }
  }

#pragma unroll
  for (int mt = 0; mt < 4; ++mt)
#pragma unroll
    for (int nt = 0; nt < 2; ++nt) {
      int col = n0 + 16 * nt + lm;
      int hh = col >> 6, d = col & 63;
#pragma unroll
      for (int r = 0; r < 8; ++r) {
        int row = m0 + 16 * mt + r + 8 * h;       // C/D layout: lanes16-31 -> M+8
        int bb = row >> 11, s = row & (SS - 1);
        unsigned short bv = bf16one(acc[mt][nt][r]);
        if (mat == 2)
          vhT[(((size_t)bb * NH + hh) * DK + d) * SS + s] = bv;
        else {
          unsigned short* dst = (mat == 0) ? qh : kh;
          dst[(((size_t)bb * NH + hh) * SS + s) * DK + d] = bv;
        }
      }
    }
}

// ---------------------------------------------------------------------------
// K2: scores = (Q @ K^T) / 8, masked -> logits in attn buffer (f32).
// Per (b,h): M=N=2048, K=64. Block 128x128, wave 64x32.
// ---------------------------------------------------------------------------
__global__ void __launch_bounds__(256, 1)
k_scores(const unsigned short* qh, const unsigned short* kh,
         const unsigned char* mask, float* attn) {
  int tid = threadIdx.x, w = tid >> 5, lane = tid & 31;
  int h = lane >> 4, lm = lane & 15;
  int bh = blockIdx.z, b = bh >> 3;
  const unsigned short* qb = qh + (size_t)bh * SS * DK;
  const unsigned short* kb = kh + (size_t)bh * SS * DK;
  int m0 = blockIdx.x * 128 + (w & 1) * 64;
  int n0 = blockIdx.y * 128 + (w >> 1) * 32;

  f32x8 zero = {0.f, 0.f, 0.f, 0.f, 0.f, 0.f, 0.f, 0.f};
  f32x8 acc[4][2];
#pragma unroll
  for (int i = 0; i < 4; ++i)
#pragma unroll
    for (int j = 0; j < 2; ++j) acc[i][j] = zero;

#pragma unroll
  for (int kk = 0; kk < DK; kk += 32) {
    bf16x16 afr[4], bfr[2];
#pragma unroll
    for (int mt = 0; mt < 4; ++mt)
      afr[mt] = load_a_bf16(qb + (size_t)(m0 + 16 * mt + lm) * DK, kk, h);
#pragma unroll
    for (int nt = 0; nt < 2; ++nt)
      bfr[nt] = load_b_bf16(kb + (size_t)(n0 + 16 * nt + lm) * DK, kk, h);
#pragma unroll
    for (int mt = 0; mt < 4; ++mt)
#pragma unroll
      for (int nt = 0; nt < 2; ++nt) acc[mt][nt] = wmma_bf16(afr[mt], bfr[nt], acc[mt][nt]);
  }

  float* arow = attn + (size_t)bh * SS * SS;
  const unsigned char* mrow = mask + (size_t)b * SS * SS;
#pragma unroll
  for (int mt = 0; mt < 4; ++mt)
#pragma unroll
    for (int nt = 0; nt < 2; ++nt) {
      int col = n0 + 16 * nt + lm;
#pragma unroll
      for (int r = 0; r < 8; ++r) {
        int row = m0 + 16 * mt + r + 8 * h;
        float val = acc[mt][nt][r] * 0.125f;   // 1/sqrt(64)
        if (mrow[(size_t)row * SS + col]) val = -1e9f;
        arow[(size_t)row * SS + col] = val;
      }
    }
}

// ---------------------------------------------------------------------------
// K3: row softmax over 2048 logits (in place). One 256-thread block per row.
// ---------------------------------------------------------------------------
__global__ void k_softmax(float* attn) {
  __shared__ float red[256];
  int tid = threadIdx.x;
  size_t row = (size_t)blockIdx.y * SS + blockIdx.x;
  float* p = attn + row * SS;

  float x[8];
  float m = -3.4e38f;
#pragma unroll
  for (int i = 0; i < 8; ++i) { x[i] = p[tid + 256 * i]; m = fmaxf(m, x[i]); }
  red[tid] = m; __syncthreads();
  for (int s = 128; s > 0; s >>= 1) {
    if (tid < s) red[tid] = fmaxf(red[tid], red[tid + s]);
    __syncthreads();
  }
  m = red[0]; __syncthreads();

  float sum = 0.f;
#pragma unroll
  for (int i = 0; i < 8; ++i) { x[i] = __expf(x[i] - m); sum += x[i]; }
  red[tid] = sum; __syncthreads();
  for (int s = 128; s > 0; s >>= 1) {
    if (tid < s) red[tid] += red[tid + s];
    __syncthreads();
  }
  float inv = 1.0f / red[0];
#pragma unroll
  for (int i = 0; i < 8; ++i) p[tid + 256 * i] = x[i] * inv;
}

// ---------------------------------------------------------------------------
// K4: heads = attn @ V. Per (b,h): M=2048, N=64, K=2048.
// Block tile 128x64, 8 waves (4 along M, 2 along N), wave tile 32x32.
// A = attn f32 (cvt->bf16), B = vhT bf16 (K-contiguous rows).
// ---------------------------------------------------------------------------
__global__ void __launch_bounds__(256, 1)
k_heads(const float* attn, const unsigned short* vhT, float* heads) {
  int tid = threadIdx.x, w = tid >> 5, lane = tid & 31;
  int h = lane >> 4, lm = lane & 15;
  int bh = blockIdx.z, b = bh >> 3, hh = bh & 7;
  int m0 = blockIdx.x * 128 + (w & 3) * 32;
  int n0 = (w >> 2) * 32;
  const float* ab = attn + (size_t)bh * SS * SS;
  const unsigned short* vb = vhT + (size_t)bh * DK * SS;

  const float* arowp[2];
  const unsigned short* vrow[2];
#pragma unroll
  for (int mt = 0; mt < 2; ++mt) arowp[mt] = ab + (size_t)(m0 + 16 * mt + lm) * SS;
#pragma unroll
  for (int nt = 0; nt < 2; ++nt) vrow[nt] = vb + (size_t)(n0 + 16 * nt + lm) * SS;

  f32x8 zero = {0.f, 0.f, 0.f, 0.f, 0.f, 0.f, 0.f, 0.f};
  f32x8 acc[2][2];
#pragma unroll
  for (int i = 0; i < 2; ++i)
#pragma unroll
    for (int j = 0; j < 2; ++j) acc[i][j] = zero;

  for (int kk = 0; kk < SS; kk += 32) {
    RawA ra[2];
#pragma unroll
    for (int mt = 0; mt < 2; ++mt) {
      if (kk + 32 < SS) __builtin_prefetch(arowp[mt] + kk + 32, 0, 0);  // global_prefetch_b8
      ra[mt] = load_a_raw(arowp[mt], kk, h);
    }
    bf16x16 bfr[2];
#pragma unroll
    for (int nt = 0; nt < 2; ++nt) bfr[nt] = load_b_bf16(vrow[nt], kk, h);
#pragma unroll
    for (int mt = 0; mt < 2; ++mt) {
      bf16x16 afr = cvt_a(ra[mt]);
#pragma unroll
      for (int nt = 0; nt < 2; ++nt) acc[mt][nt] = wmma_bf16(afr, bfr[nt], acc[mt][nt]);
    }
  }

#pragma unroll
  for (int mt = 0; mt < 2; ++mt)
#pragma unroll
    for (int nt = 0; nt < 2; ++nt) {
      int d = n0 + 16 * nt + lm;
#pragma unroll
      for (int r = 0; r < 8; ++r) {
        int row = m0 + 16 * mt + r + 8 * h;
        heads[((size_t)b * SS + row) * DM + hh * DK + d] = acc[mt][nt][r];
      }
    }
}

// ---------------------------------------------------------------------------
// K5: preout = heads @ W_o + residual(q). M=8192, N=512, K=512.
// Same tiling as K1; writes f32 into d_out[0 .. B*S*DM).
// ---------------------------------------------------------------------------
__global__ void __launch_bounds__(256, 1)
k_out(const float* heads, const unsigned short* woT,
      const float* resid, float* out) {
  int tid = threadIdx.x, w = tid >> 5, lane = tid & 31;
  int h = lane >> 4, lm = lane & 15;
  int m0 = blockIdx.x * 128 + (w & 1) * 64;
  int n0 = blockIdx.y * 128 + (w >> 1) * 32;

  const float* xrow[4];
  const unsigned short* wrow[2];
#pragma unroll
  for (int mt = 0; mt < 4; ++mt) xrow[mt] = heads + (size_t)(m0 + 16 * mt + lm) * DM;
#pragma unroll
  for (int nt = 0; nt < 2; ++nt) wrow[nt] = woT + (size_t)(n0 + 16 * nt + lm) * DM;

  f32x8 zero = {0.f, 0.f, 0.f, 0.f, 0.f, 0.f, 0.f, 0.f};
  f32x8 acc[4][2];
#pragma unroll
  for (int i = 0; i < 4; ++i)
#pragma unroll
    for (int j = 0; j < 2; ++j) acc[i][j] = zero;

  for (int kk = 0; kk < DM; kk += 32) {
    RawA ra[4];
#pragma unroll
    for (int mt = 0; mt < 4; ++mt) ra[mt] = load_a_raw(xrow[mt], kk, h);
    bf16x16 bfr[2];
#pragma unroll
    for (int nt = 0; nt < 2; ++nt) bfr[nt] = load_b_bf16(wrow[nt], kk, h);
#pragma unroll
    for (int mt = 0; mt < 4; ++mt) {
      bf16x16 afr = cvt_a(ra[mt]);
#pragma unroll
      for (int nt = 0; nt < 2; ++nt) acc[mt][nt] = wmma_bf16(afr, bfr[nt], acc[mt][nt]);
    }
  }

#pragma unroll
  for (int mt = 0; mt < 4; ++mt)
#pragma unroll
    for (int nt = 0; nt < 2; ++nt) {
      int col = n0 + 16 * nt + lm;
#pragma unroll
      for (int r = 0; r < 8; ++r) {
        int row = m0 + 16 * mt + r + 8 * h;
        out[(size_t)row * DM + col] = acc[mt][nt][r] + resid[(size_t)row * DM + col];
      }
    }
}

// ---------------------------------------------------------------------------
// K6: LayerNorm over rows of 512, in place on d_out. One block per row.
// ---------------------------------------------------------------------------
__global__ void k_ln(float* out, const float* g, const float* bta) {
  __shared__ float red[256];
  int tid = threadIdx.x;
  float* p = out + (size_t)blockIdx.x * DM;
  float x0 = p[tid], x1 = p[tid + 256];

  red[tid] = x0 + x1; __syncthreads();
  for (int s = 128; s > 0; s >>= 1) {
    if (tid < s) red[tid] += red[tid + s];
    __syncthreads();
  }
  float mu = red[0] * (1.0f / DM); __syncthreads();

  float d0 = x0 - mu, d1 = x1 - mu;
  red[tid] = d0 * d0 + d1 * d1; __syncthreads();
  for (int s = 128; s > 0; s >>= 1) {
    if (tid < s) red[tid] += red[tid + s];
    __syncthreads();
  }
  float rs = rsqrtf(red[0] * (1.0f / DM) + EPSLN);

  p[tid]       = d0 * rs * g[tid]       + bta[tid];
  p[tid + 256] = d1 * rs * g[tid + 256] + bta[tid + 256];
}

// ---------------------------------------------------------------------------
extern "C" void kernel_launch(void* const* d_in, const int* in_sizes, int n_in,
                              void* d_out, int out_size, void* d_ws, size_t ws_size,
                              hipStream_t stream) {
  (void)in_sizes; (void)n_in; (void)out_size; (void)ws_size;

  const float* q    = (const float*)d_in[0];
  const float* k    = (const float*)d_in[1];
  const float* v    = (const float*)d_in[2];
  const float* w_q  = (const float*)d_in[3];
  const float* w_k  = (const float*)d_in[4];
  const float* w_v  = (const float*)d_in[5];
  const float* w_o  = (const float*)d_in[6];
  const float* ln_g = (const float*)d_in[7];
  const float* ln_b = (const float*)d_in[8];
  const unsigned char* mask = (const unsigned char*)d_in[9];  // jax bool_: 1 byte

  char* ws = (char*)d_ws;
  unsigned short* qh   = (unsigned short*)(ws);                              // 8 MB
  unsigned short* kh   = (unsigned short*)(ws + ((size_t)8  << 20));         // 8 MB
  unsigned short* vhT  = (unsigned short*)(ws + ((size_t)16 << 20));         // 8 MB
  float*          heads = (float*)        (ws + ((size_t)24 << 20));         // 16 MB
  unsigned short* wqT  = (unsigned short*)(ws + ((size_t)40 << 20));         // 512 KB
  unsigned short* wkT  = (unsigned short*)(ws + ((size_t)40 << 20) + (512u << 10));
  unsigned short* wvT  = (unsigned short*)(ws + ((size_t)41 << 20));
  unsigned short* woT  = (unsigned short*)(ws + ((size_t)41 << 20) + (512u << 10));

  float* out  = (float*)d_out;
  float* attn = out + (size_t)BB * SS * DM;

  k_wtrans <<<dim3(1024, 4),    256, 0, stream>>>(w_q, w_k, w_v, w_o, wqT, wkT, wvT, woT);
  k_proj   <<<dim3(64, 4, 3),   256, 0, stream>>>(q, k, v, wqT, wkT, wvT, qh, kh, vhT);
  k_scores <<<dim3(16, 16, 32), 256, 0, stream>>>(qh, kh, mask, attn);
  k_softmax<<<dim3(SS, 32),     256, 0, stream>>>(attn);
  k_heads  <<<dim3(16, 1, 32),  256, 0, stream>>>(attn, vhT, heads);
  k_out    <<<dim3(64, 4),      256, 0, stream>>>(heads, woT, q, out);
  k_ln     <<<dim3(BB * SS),    256, 0, stream>>>(out, ln_g, ln_b);
}
